// LlamaAttention_14989435863727
// MI455X (gfx1250) — matrix operand order repaired and checked by reading, not compiled, not run on hardware
//
#include <hip/hip_runtime.h>
#include <stdint.h>

typedef __bf16 bf16;
typedef uint32_t u32;
typedef __attribute__((ext_vector_type(16))) bf16  v16bf;
typedef __attribute__((ext_vector_type(8)))  float v8f;

#define N_HEADS   32
#define N_KVHEADS 8
#define HD        64
#define BB        2
#define TT        2048
#define DMODEL    2048
#define KVDIM     512
#define QKVN      3072   // 2048 + 512 + 512

// ---------------------------------------------------------------------------
// Fragment loaders, matching CDNA5 WMMA VGPR layouts (wave32).
// A (16-bit 16x32): lanes 0-15 hold K=[0,8)+[16,24); lanes 16-31 K=[8,16)+[24,32)
// B (16-bit 32x16): lanes 0-15 hold K=[0,8)+[16,24); lanes 16-31 K=[8,16)+[24,32)
// rowp points at 32 contiguous bf16 (16 u32) of the lane's row, 16B-aligned.
// ---------------------------------------------------------------------------
union FragCast { uint4 q[2]; v16bf v; };

static __device__ __forceinline__ v16bf ld_frag_a(const u32* rowp, int half) {
  FragCast f;
  f.q[0] = *(const uint4*)(rowp + half * 4);
  f.q[1] = *(const uint4*)(rowp + 8 + half * 4);
  return f.v;
}
static __device__ __forceinline__ v16bf ld_frag_b(const u32* rowp, int half) {
  FragCast f;
  f.q[0] = *(const uint4*)(rowp + half * 4);
  f.q[1] = *(const uint4*)(rowp + 8 + half * 4);
  return f.v;
}

// ---------------------------------------------------------------------------
// CDNA5 async global->LDS copies (ASYNCcnt-tracked, VGPR-bypassing).
// INST_OFFSET is applied to BOTH the global and LDS addresses, so a single
// (lds, global) base pair stages a contiguous chunk.
// ---------------------------------------------------------------------------
static __device__ __forceinline__ void async_copy64B(u32 lds_off, const void* g) {
  asm volatile(
      "global_load_async_to_lds_b128 %0, %1, off\n\t"
      "global_load_async_to_lds_b128 %0, %1, off offset:16\n\t"
      "global_load_async_to_lds_b128 %0, %1, off offset:32\n\t"
      "global_load_async_to_lds_b128 %0, %1, off offset:48"
      :: "v"(lds_off), "v"((uint64_t)(uintptr_t)g) : "memory");
}
static __device__ __forceinline__ void async_copy32B(u32 lds_off, const void* g) {
  asm volatile(
      "global_load_async_to_lds_b128 %0, %1, off\n\t"
      "global_load_async_to_lds_b128 %0, %1, off offset:16"
      :: "v"(lds_off), "v"((uint64_t)(uintptr_t)g) : "memory");
}
static __device__ __forceinline__ u32 lds_off32(const void* p) {
  return (u32)(uintptr_t)p;   // flat LDS address low 32 bits = wave-relative offset
}

// ---------------------------------------------------------------------------
// Kernel 1a: f32 -> bf16 elementwise cast
// ---------------------------------------------------------------------------
__global__ __launch_bounds__(256) void cast_bf16_kernel(
    const float* __restrict__ src, bf16* __restrict__ dst, int n) {
  int i = blockIdx.x * 256 + threadIdx.x;
  if (i < n) dst[i] = (bf16)src[i];
}

// ---------------------------------------------------------------------------
// Kernel 1b: f32 [K][N] -> bf16 [N][K] transposed cast (weights)
// ---------------------------------------------------------------------------
__global__ __launch_bounds__(256) void transpose_bf16_kernel(
    const float* __restrict__ src, bf16* __restrict__ dst, int K, int N) {
  int i = blockIdx.x * 256 + threadIdx.x;
  if (i >= K * N) return;
  int k = i / N, n = i % N;
  dst[(size_t)n * K + k] = (bf16)src[i];
}

// ---------------------------------------------------------------------------
// Kernel 2/5: C[M,N](f32) = A[M,K](bf16, row-major) @ BT[N,K](bf16)^T
// 256 threads = 8 waves, tile 128x128, K-step 64, async double-buffered LDS.
// Wave grid 4(M) x 2(N): each wave -> 32x64 = 2x4 WMMA C tiles.
// ---------------------------------------------------------------------------
__global__ __launch_bounds__(256) void gemm_bf16_wmma_kernel(
    const bf16* __restrict__ A, const bf16* __restrict__ BT,
    float* __restrict__ C, int M, int N, int K) {
  __shared__ __attribute__((aligned(16))) u32 As[2][128][36];  // 64 bf16 + pad
  __shared__ __attribute__((aligned(16))) u32 Bs[2][128][36];

  const int tid  = threadIdx.x;
  const int lane = tid & 31;
  const int w    = tid >> 5;
  const int l16  = lane & 15;
  const int half = lane >> 4;
  const int m0 = blockIdx.y * 128;
  const int n0 = blockIdx.x * 128;
  const int wm = (w & 3) * 32;
  const int wn = (w >> 2) * 64;

  v8f acc[2][4];
#pragma unroll
  for (int mt = 0; mt < 2; ++mt)
#pragma unroll
    for (int nt = 0; nt < 4; ++nt)
#pragma unroll
      for (int r = 0; r < 8; ++r) acc[mt][nt][r] = 0.0f;

  const int srow = tid >> 1;
  const int sseg = (tid & 1) * 16;      // u32 offset inside 32-u32 k-row
  const int Ku   = K >> 1;              // u32 per full row
  const u32* Ag = (const u32*)A  + (size_t)(m0 + srow) * Ku + sseg;
  const u32* Bg = (const u32*)BT + (size_t)(n0 + srow) * Ku + sseg;

  // prologue: stage k-block 0 into buffer 0 (8 async b128 per thread)
  async_copy64B(lds_off32(&As[0][srow][sseg]), Ag);
  async_copy64B(lds_off32(&Bs[0][srow][sseg]), Bg);

  const int nk = K >> 6;
  for (int ki = 0; ki < nk; ++ki) {
    const int buf = ki & 1;
    if (ki + 1 < nk) {
      // stage next k-block into the other buffer, then retire current stage
      async_copy64B(lds_off32(&As[buf ^ 1][srow][sseg]), Ag + ((ki + 1) << 5));
      async_copy64B(lds_off32(&Bs[buf ^ 1][srow][sseg]), Bg + ((ki + 1) << 5));
      asm volatile("s_wait_asynccnt 0x8" ::: "memory");
    } else {
      asm volatile("s_wait_asynccnt 0x0" ::: "memory");
    }
    __syncthreads();

#pragma unroll
    for (int kk = 0; kk < 2; ++kk) {
      v16bf af[2], bfv[4];
#pragma unroll
      for (int mt = 0; mt < 2; ++mt)
        af[mt] = ld_frag_a(&As[buf][wm + mt * 16 + l16][kk * 16], half);
#pragma unroll
      for (int nt = 0; nt < 4; ++nt)
        bfv[nt] = ld_frag_b(&Bs[buf][wn + nt * 16 + l16][kk * 16], half);
#pragma unroll
      for (int mt = 0; mt < 2; ++mt)
#pragma unroll
        for (int nt = 0; nt < 4; ++nt)
          acc[mt][nt] = __builtin_amdgcn_wmma_f32_16x16x32_bf16(
              false, af[mt], false, bfv[nt], (short)0, acc[mt][nt], false, false);
    }
    __syncthreads();
  }

  // C tile layout: lanes 0-15 -> M=r, lanes 16-31 -> M=r+8; N=l16
#pragma unroll
  for (int mt = 0; mt < 2; ++mt)
#pragma unroll
    for (int nt = 0; nt < 4; ++nt)
#pragma unroll
      for (int r = 0; r < 8; ++r) {
        const int gm = m0 + wm + mt * 16 + half * 8 + r;
        const int gn = n0 + wn + nt * 16 + l16;
        C[(size_t)gm * N + gn] = acc[mt][nt][r];
      }
}

// ---------------------------------------------------------------------------
// Kernel 3: RoPE + QKV split/layout.
//   qkv f32 [B*T][3072] ->
//   Q bf16 [B][32][T][64] (rope), K bf16 [B][8][T][64] (rope),
//   VT bf16 [B][8][64][T] (transposed so flash-attn B-fragments are contiguous)
// ---------------------------------------------------------------------------
__global__ __launch_bounds__(256) void rope_split_kernel(
    const float* __restrict__ qkv, bf16* __restrict__ Q,
    bf16* __restrict__ K, bf16* __restrict__ VT) {
  const int i = blockIdx.x * 256 + threadIdx.x;   // over 4096*3072
  const int row = i / QKVN;                       // b*T + t
  const int col = i % QKVN;
  const int b = row >> 11;
  const int t = row & (TT - 1);
  const float val = qkv[(size_t)row * QKVN + col];

  if (col < DMODEL) {                             // Q path (rope)
    const int h = col >> 6, d = col & 63, j = d >> 1;
    const float pv  = qkv[(size_t)row * QKVN + (col ^ 1)];
    const float inv = __expf(-0.14391157f * (float)(2 * j));  // theta^(-2j/64)
    float s, c;
    __sincosf((float)t * inv, &s, &c);
    const float o = (d & 1) ? (pv * s + val * c) : (val * c - pv * s);
    Q[((size_t)(b * N_HEADS + h) * TT + t) * HD + d] = (bf16)o;
  } else if (col < DMODEL + KVDIM) {              // K path (rope)
    const int cc = col - DMODEL;
    const int kvh = cc >> 6, d = cc & 63, j = d >> 1;
    const float pv  = qkv[(size_t)row * QKVN + (col ^ 1)];
    const float inv = __expf(-0.14391157f * (float)(2 * j));
    float s, c;
    __sincosf((float)t * inv, &s, &c);
    const float o = (d & 1) ? (pv * s + val * c) : (val * c - pv * s);
    K[((size_t)(b * N_KVHEADS + kvh) * TT + t) * HD + d] = (bf16)o;
  } else {                                        // V path (transposed store)
    const int cc = col - DMODEL - KVDIM;
    const int kvh = cc >> 6, d = cc & 63;
    VT[((size_t)(b * N_KVHEADS + kvh) * HD + d) * TT + t] = (bf16)val;
  }
}

// ---------------------------------------------------------------------------
// Kernel 4: causal flash attention (bf16 WMMA, f32 online softmax).
// grid = (T/128, B*32). 8 waves/block, each wave owns 16 query rows.
// K/V blocks staged once per workgroup via async double-buffered LDS.
// ---------------------------------------------------------------------------
__global__ __launch_bounds__(256) void flash_attn_kernel(
    const bf16* __restrict__ Q, const bf16* __restrict__ K,
    const bf16* __restrict__ VT, bf16* __restrict__ O) {
  __shared__ __attribute__((aligned(16))) u32  Ks[2][64][36];  // [tok][d]
  __shared__ __attribute__((aligned(16))) u32  Vs[2][64][36];  // [d][tok]
  __shared__ __attribute__((aligned(16))) bf16 Ps[8][16][72];  // per-wave P

  const int tid  = threadIdx.x;
  const int lane = tid & 31;
  const int w    = tid >> 5;
  const int l16  = lane & 15;
  const int half = lane >> 4;
  const int bh = blockIdx.y;
  const int b  = bh >> 5;
  const int h  = bh & 31;
  const int kvh = h >> 2;                         // GQA group of 4
  const int qW = blockIdx.x * 128 + w * 16;       // wave's first query

  const bf16* qbase = Q  + (size_t)(b * N_HEADS   + h)   * TT * HD;
  const bf16* kbase = K  + (size_t)(b * N_KVHEADS + kvh) * TT * HD;
  const bf16* vbase = VT + (size_t)(b * N_KVHEADS + kvh) * HD * TT;

  // Q A-fragments for the two K=32 slices of head_dim=64 (kept in regs)
  v16bf qa[2];
  {
    const u32* qp = (const u32*)(qbase + (size_t)(qW + l16) * HD);
    qa[0] = ld_frag_a(qp, half);
    qa[1] = ld_frag_a(qp + 16, half);
  }

  v8f   o_acc[4];
  float m_i[8], l_i[8];
#pragma unroll
  for (int dt = 0; dt < 4; ++dt)
#pragma unroll
    for (int r = 0; r < 8; ++r) o_acc[dt][r] = 0.0f;
#pragma unroll
  for (int r = 0; r < 8; ++r) { m_i[r] = -1e30f; l_i[r] = 0.0f; }

  // staging coordinates: thread -> (row 0..63, 32B segment 0..3)
  const int krow = tid >> 2;
  const int kseg = (tid & 3) * 8;                     // u32
  const u32* kg = (const u32*)kbase + (size_t)krow * (HD / 2) + kseg;
  const u32* vg = (const u32*)vbase + (size_t)krow * (TT / 2) + kseg;

  // block-uniform causal bound: keys < (blockIdx.x+1)*128, never OOB
  const int nkb = 2 * (blockIdx.x + 1);

  // prologue: stage key-block 0 (2 K + 2 V async b128 per thread)
  async_copy32B(lds_off32(&Ks[0][krow][kseg]), kg);
  async_copy32B(lds_off32(&Vs[0][krow][kseg]), vg);

  for (int kbI = 0; kbI < nkb; ++kbI) {
    const int kb  = kbI * 64;
    const int buf = kbI & 1;
    if (kbI + 1 < nkb) {
      async_copy32B(lds_off32(&Ks[buf ^ 1][krow][kseg]), kg + (size_t)(kbI + 1) * 64 * (HD / 2));
      async_copy32B(lds_off32(&Vs[buf ^ 1][krow][kseg]), vg + (size_t)(kbI + 1) * 32);
      asm volatile("s_wait_asynccnt 0x4" ::: "memory");
    } else {
      asm volatile("s_wait_asynccnt 0x0" ::: "memory");
    }
    __syncthreads();

    if (kb <= qW + 15) {   // wave-uniform causal skip
      // ---- S = Q @ K^T ----
      v8f s[4];
#pragma unroll
      for (int nt = 0; nt < 4; ++nt)
#pragma unroll
        for (int r = 0; r < 8; ++r) s[nt][r] = 0.0f;

#pragma unroll
      for (int kk = 0; kk < 2; ++kk)
#pragma unroll
        for (int nt = 0; nt < 4; ++nt) {
          v16bf kf = ld_frag_b(&Ks[buf][nt * 16 + l16][kk * 16], half);
          s[nt] = __builtin_amdgcn_wmma_f32_16x16x32_bf16(
              false, qa[kk], false, kf, (short)0, s[nt], false, false);
        }

      // ---- scale + causal mask ----
#pragma unroll
      for (int nt = 0; nt < 4; ++nt) {
        const int tk = kb + nt * 16 + l16;
#pragma unroll
        for (int r = 0; r < 8; ++r) {
          const int tq = qW + half * 8 + r;
          const float v = s[nt][r] * 0.125f;      // 1/sqrt(64)
          s[nt][r] = (tk > tq) ? -1e30f : v;
        }
      }

      // ---- online softmax (rows live in 16-lane halves -> xor 1,2,4,8) ----
      float alpha[8];
#pragma unroll
      for (int r = 0; r < 8; ++r) {
        float rm = s[0][r];
#pragma unroll
        for (int nt = 1; nt < 4; ++nt) rm = fmaxf(rm, s[nt][r]);
#pragma unroll
        for (int off = 1; off < 16; off <<= 1)
          rm = fmaxf(rm, __shfl_xor(rm, off, 32));
        const float mn = fmaxf(m_i[r], rm);
        alpha[r] = __expf(m_i[r] - mn);
        m_i[r] = mn;
        float rs = 0.0f;
#pragma unroll
        for (int nt = 0; nt < 4; ++nt) {
          const float p = __expf(s[nt][r] - mn);
          s[nt][r] = p;
          rs += p;
        }
#pragma unroll
        for (int off = 1; off < 16; off <<= 1)
          rs += __shfl_xor(rs, off, 32);
        l_i[r] = l_i[r] * alpha[r] + rs;
      }
#pragma unroll
      for (int dt = 0; dt < 4; ++dt)
#pragma unroll
        for (int r = 0; r < 8; ++r) o_acc[dt][r] *= alpha[r];

      // ---- P (C-layout regs) -> LDS, re-read as A-fragment ----
#pragma unroll
      for (int nt = 0; nt < 4; ++nt)
#pragma unroll
        for (int r = 0; r < 8; ++r)
          Ps[w][half * 8 + r][nt * 16 + l16] = (bf16)s[nt][r];

      // ---- O += P @ V ----
      const u32* pp = (const u32*)&Ps[w][l16][0];
#pragma unroll
      for (int kk = 0; kk < 2; ++kk) {
        v16bf pa = ld_frag_a(pp + kk * 16, half);
#pragma unroll
        for (int dt = 0; dt < 4; ++dt) {
          v16bf vf = ld_frag_b(&Vs[buf][dt * 16 + l16][kk * 16], half);
          o_acc[dt] = __builtin_amdgcn_wmma_f32_16x16x32_bf16(
              false, pa, false, vf, (short)0, o_acc[dt], false, false);
        }
      }
    }
    __syncthreads();
  }

  // ---- normalize and store O as bf16 [B*T][2048] for the out-projection ----
#pragma unroll
  for (int dt = 0; dt < 4; ++dt)
#pragma unroll
    for (int r = 0; r < 8; ++r) {
      const int t = qW + half * 8 + r;
      const float ov = o_acc[dt][r] / l_i[r];
      O[(size_t)(b * TT + t) * DMODEL + h * 64 + dt * 16 + l16] = (bf16)ov;
    }
}

// ---------------------------------------------------------------------------
// Host-side orchestration
// ---------------------------------------------------------------------------
extern "C" void kernel_launch(void* const* d_in, const int* in_sizes, int n_in,
                              void* d_out, int out_size, void* d_ws, size_t ws_size,
                              hipStream_t stream) {
  const float* x  = (const float*)d_in[0];
  const float* wq = (const float*)d_in[1];
  const float* wk = (const float*)d_in[2];
  const float* wv = (const float*)d_in[3];
  const float* wo = (const float*)d_in[4];

  const size_t BT = (size_t)BB * TT;              // 4096 rows

  // workspace carve-up (~124 MiB total)
  char* ws = (char*)d_ws;
  size_t off = 0;
  bf16*  xb    = (bf16*)(ws + off); off += BT * DMODEL * 2;                 // 16M
  bf16*  wqkvT = (bf16*)(ws + off); off += (size_t)QKVN * DMODEL * 2;       // 12M
  bf16*  woT   = (bf16*)(ws + off); off += (size_t)DMODEL * DMODEL * 2;     // 8M
  float* qkvf  = (float*)(ws + off); off += BT * QKVN * 4;                  // 48M
  bf16*  Qb    = (bf16*)(ws + off); off += BT * DMODEL * 2;                 // 16M
  bf16*  Kb    = (bf16*)(ws + off); off += BT * KVDIM * 2;                  // 4M
  bf16*  VTb   = (bf16*)(ws + off); off += BT * KVDIM * 2;                  // 4M
  bf16*  Ob    = (bf16*)(ws + off); off += BT * DMODEL * 2;                 // 16M
  if (off > ws_size) return;  // deterministic no-op if scratch too small

  // 1) casts / weight transposes
  {
    const int nx = (int)(BT * DMODEL);
    cast_bf16_kernel<<<nx / 256, 256, 0, stream>>>(x, xb, nx);
    const int nw = DMODEL * DMODEL;
    transpose_bf16_kernel<<<nw / 256, 256, 0, stream>>>(wq, wqkvT, DMODEL, DMODEL);
    const int nk = DMODEL * KVDIM;
    transpose_bf16_kernel<<<nk / 256, 256, 0, stream>>>(
        wk, wqkvT + (size_t)DMODEL * DMODEL, DMODEL, KVDIM);
    transpose_bf16_kernel<<<nk / 256, 256, 0, stream>>>(
        wv, wqkvT + (size_t)(DMODEL + KVDIM) * DMODEL, DMODEL, KVDIM);
    transpose_bf16_kernel<<<nw / 256, 256, 0, stream>>>(wo, woT, DMODEL, DMODEL);
  }

  // 2) fused QKV projection: [4096,2048] x [2048,3072] -> f32
  gemm_bf16_wmma_kernel<<<dim3(QKVN / 128, (int)(BT / 128)), 256, 0, stream>>>(
      xb, wqkvT, qkvf, (int)BT, QKVN, DMODEL);

  // 3) RoPE + head split (+ V transpose)
  {
    const int n = (int)(BT * QKVN);
    rope_split_kernel<<<n / 256, 256, 0, stream>>>(qkvf, Qb, Kb, VTb);
  }

  // 4) causal flash attention
  flash_attn_kernel<<<dim3(TT / 128, BB * N_HEADS), 256, 0, stream>>>(
      Qb, Kb, VTb, Ob);

  // 5) output projection -> f32 d_out
  gemm_bf16_wmma_kernel<<<dim3(DMODEL / 128, (int)(BT / 128)), 256, 0, stream>>>(
      Ob, woT, (float*)d_out, (int)BT, DMODEL, DMODEL);
}
